// build_model_75728863363698
// MI455X (gfx1250) — matrix operand order repaired and checked by reading, not compiled, not run on hardware
//
#include <hip/hip_runtime.h>

// Problem constants (from reference): B=256, T=64, I=512, H=1024, O=1
#define B_   256
#define T_   64
#define I_   512
#define H_   1024
#define G3_  3072      // 3*H
#define BT_  16384     // B*T

typedef __attribute__((ext_vector_type(8)))  __bf16 v8bf;
typedef __attribute__((ext_vector_type(16))) __bf16 v16bf;
typedef __attribute__((ext_vector_type(8)))  float  v8f;

__device__ __forceinline__ float sigmoidf_(float x) { return 1.0f / (1.0f + __expf(-x)); }

__device__ __forceinline__ v16bf cat8(v8bf lo, v8bf hi) {
  return __builtin_shufflevector(lo, hi, 0,1,2,3,4,5,6,7,8,9,10,11,12,13,14,15);
}

// ---------------------------------------------------------------------------
// 8-wave (256-thread, wave32) cooperative GEMM core, templated on per-wave
// row-tiling WM (1 or 2). Block computes a (WM*64) x 64 tile of
//   C = A[M,K](bf16,row-major,lda) x W[N,K]^T   (PyTorch weight layout).
// Wave wv: rows (wv&3)*WM*16 .. +WM*16-1, cols (wv>>2)*32 .. +31
//   -> WM x 2 accumulators of 16x16, WM*2 WMMAs per k-step off WM+2 fragments.
// Fragment layouts per CDNA5 ISA:
//   A 16x32 bf16 : lane(l) row = l&15; elem j <-> k = (j>>3)*16 + (l>>4)*8 + (j&7)
//   B 32x16 bf16 : lane(l) col = l&15; elem j <-> k = (j>>3)*16 + (l>>4)*8 + (j&7)
//   C/D 16x16 f32: vgpr r -> row = r + (l>>4)*8, col = l&15
// ---------------------------------------------------------------------------
template <int WM>
__device__ __forceinline__ void gemm_core(const __bf16* __restrict__ A,
                                          const __bf16* __restrict__ W,
                                          int K, int lda, int ldb,
                                          int m0, int n0,
                                          __bf16* Abuf,   // [WM*64*32]
                                          __bf16* Wbuf,   // [64*32]
                                          v8f (&acc)[WM][2])
{
  const int tid  = threadIdx.x;
  const int lane = tid & 31;
  const int wv   = tid >> 5;
  const int wm   = wv & 3;         // wave row (0..3)
  const int wn   = wv >> 2;        // wave col (0..1)
  const int lrow = lane & 15;
  const int half = lane >> 4;

  for (int k0 = 0; k0 < K; k0 += 32) {
    __syncthreads();  // protect LDS from previous iteration's readers
#pragma unroll
    for (int i = 0; i < WM; ++i) {          // A tile: WM*64 rows x 32 cols
      const int c = tid + i * 256;          // 16B chunk id
      const int r = c >> 2, cc = (c & 3) * 8;
      *(v8bf*)(Abuf + r * 32 + cc) =
          *(const v8bf*)(A + (size_t)(m0 + r) * lda + k0 + cc);
    }
    {
      const int r = tid >> 2, cc = (tid & 3) * 8;   // W tile: 64 x 32
      *(v8bf*)(Wbuf + r * 32 + cc) =
          *(const v8bf*)(W + (size_t)(n0 + r) * ldb + k0 + cc);
      if (k0 + 32 < K) {
        __builtin_prefetch(W + (size_t)(n0 + r) * ldb + k0 + 32 + cc, 0, 1);
#pragma unroll
        for (int i = 0; i < WM; ++i) {
          const int c2 = tid + i * 256;
          __builtin_prefetch(
              A + (size_t)(m0 + (c2 >> 2)) * lda + k0 + 32 + (c2 & 3) * 8, 0, 1);
        }
      }
    }
    __syncthreads();

    v16bf af[WM];
#pragma unroll
    for (int m = 0; m < WM; ++m) {
      const __bf16* base = Abuf + ((wm * WM + m) * 16 + lrow) * 32 + half * 8;
      af[m] = cat8(*(const v8bf*)(base), *(const v8bf*)(base + 16));
    }
    v16bf bfr[2];
#pragma unroll
    for (int t = 0; t < 2; ++t) {
      const __bf16* base = Wbuf + (wn * 32 + t * 16 + lrow) * 32 + half * 8;
      bfr[t] = cat8(*(const v8bf*)(base), *(const v8bf*)(base + 16));
    }
#pragma unroll
    for (int m = 0; m < WM; ++m)
#pragma unroll
      for (int t = 0; t < 2; ++t)
        acc[m][t] = __builtin_amdgcn_wmma_f32_16x16x32_bf16(
            false, af[m], false, bfr[t], (short)0, acc[m][t], false, false);
  }
}

// ---------------------------------------------------------------------------
// Plain GEMM, f32 output:  C[M,N] = A x W^T   (used for gx1 and gh)
// ---------------------------------------------------------------------------
template <int WM>
__global__ __launch_bounds__(256) void k_gemm_f32(const __bf16* __restrict__ A,
                                                  const __bf16* __restrict__ W,
                                                  float* __restrict__ C,
                                                  int K, int lda, int ldb, int ldc)
{
  __shared__ __bf16 Abuf[WM * 64 * 32];
  __shared__ __bf16 Wbuf[64 * 32];
  const int bm = blockIdx.x * (WM * 64), bn = blockIdx.y * 64;
  v8f acc[WM][2] = {};
  gemm_core<WM>(A, W, K, lda, ldb, bm, bn, Abuf, Wbuf, acc);

  const int tid = threadIdx.x, lane = tid & 31, wv = tid >> 5;
  const int half = lane >> 4;
  const int col0 = bn + (wv >> 2) * 32 + (lane & 15);
#pragma unroll
  for (int m = 0; m < WM; ++m) {
    const int row0 = bm + ((wv & 3) * WM + m) * 16 + half * 8;
#pragma unroll
    for (int r = 0; r < 8; ++r) {
      C[(size_t)(row0 + r) * ldc + col0]      = acc[m][0][r];
      C[(size_t)(row0 + r) * ldc + col0 + 16] = acc[m][1][r];
    }
  }
}

// ---------------------------------------------------------------------------
// Layer 0 fused: h0 = (1 - sigmoid(a@Wz^T)) * tanh(a@Wn^T), stored bf16.
// (hidden starts at 0 each step, so the r-gate and Whh0 are dead.)
// ---------------------------------------------------------------------------
__global__ __launch_bounds__(256) void k_layer0(const __bf16* __restrict__ Abf,
                                                const __bf16* __restrict__ W0bf,
                                                __bf16* __restrict__ h0)
{
  __shared__ __bf16 Abuf[2 * 64 * 32];
  __shared__ __bf16 Wbuf[64 * 32];
  const int bm = blockIdx.x * 128, bn = blockIdx.y * 64;
  const __bf16* Wz = W0bf + (size_t)H_ * I_;       // rows [H,2H)  -> z gate
  const __bf16* Wn = W0bf + (size_t)2 * H_ * I_;   // rows [2H,3H) -> n gate

  v8f az[2][2] = {};
  v8f an[2][2] = {};
  gemm_core<2>(Abf, Wz, I_, I_, I_, bm, bn, Abuf, Wbuf, az);
  gemm_core<2>(Abf, Wn, I_, I_, I_, bm, bn, Abuf, Wbuf, an);

  const int tid = threadIdx.x, lane = tid & 31, wv = tid >> 5;
  const int half = lane >> 4;
  const int col0 = bn + (wv >> 2) * 32 + (lane & 15);
#pragma unroll
  for (int m = 0; m < 2; ++m) {
    const int row0 = bm + ((wv & 3) * 2 + m) * 16 + half * 8;
#pragma unroll
    for (int r = 0; r < 8; ++r) {
      float h0a = (1.0f - sigmoidf_(az[m][0][r])) * tanhf(an[m][0][r]);
      float h0b = (1.0f - sigmoidf_(az[m][1][r])) * tanhf(an[m][1][r]);
      h0[(size_t)(row0 + r) * H_ + col0]      = (__bf16)h0a;
      h0[(size_t)(row0 + r) * H_ + col0 + 16] = (__bf16)h0b;
    }
  }
}

// ---------------------------------------------------------------------------
// s_new = sigmoid(h @ Ws^T): write f32 carry, bf16 carry, and d_out s_list.
// ---------------------------------------------------------------------------
__global__ __launch_bounds__(256) void k_sgemm_sig(const __bf16* __restrict__ hbf,
                                                   const __bf16* __restrict__ Wsb,
                                                   float* __restrict__ s_f,
                                                   __bf16* __restrict__ s_bf,
                                                   float* __restrict__ out_s,
                                                   int t)
{
  __shared__ __bf16 Abuf[64 * 32];
  __shared__ __bf16 Wbuf[64 * 32];
  const int bm = blockIdx.x * 64, bn = blockIdx.y * 64;
  v8f acc[1][2] = {};
  gemm_core<1>(hbf, Wsb, H_, H_, H_, bm, bn, Abuf, Wbuf, acc);

  const int tid = threadIdx.x, lane = tid & 31, wv = tid >> 5;
  const int row0 = bm + (wv & 3) * 16 + (lane >> 4) * 8;
  const int col0 = bn + (wv >> 2) * 32 + (lane & 15);
#pragma unroll
  for (int r = 0; r < 8; ++r) {
    const int b = row0 + r;
    float v0 = sigmoidf_(acc[0][0][r]);
    float v1 = sigmoidf_(acc[0][1][r]);
    s_f[(size_t)b * H_ + col0]      = v0;
    s_f[(size_t)b * H_ + col0 + 16] = v1;
    s_bf[(size_t)b * H_ + col0]      = (__bf16)v0;
    s_bf[(size_t)b * H_ + col0 + 16] = (__bf16)v1;
    out_s[((size_t)b * T_ + t) * H_ + col0]      = v0;
    out_s[((size_t)b * T_ + t) * H_ + col0 + 16] = v1;
  }
}

// ---------------------------------------------------------------------------
// GRU gate math (elementwise over [B,H]); produces h (f32 + bf16 copies).
// ---------------------------------------------------------------------------
__global__ __launch_bounds__(256) void k_gates(const float* __restrict__ gx1,
                                               const float* __restrict__ gh,
                                               const float* __restrict__ s,
                                               float* __restrict__ h,
                                               __bf16* __restrict__ hbf,
                                               int t)
{
  const int i = blockIdx.x * 256 + threadIdx.x;   // 0 .. B*H-1
  const int b = i >> 10;
  const int c = i & (H_ - 1);
  const size_t gx = ((size_t)b * T_ + t) * G3_;
  const size_t gb = (size_t)b * G3_;
  float r = sigmoidf_(gx1[gx + c]          + gh[gb + c]);
  float z = sigmoidf_(gx1[gx + H_ + c]     + gh[gb + H_ + c]);
  float n = tanhf    (gx1[gx + 2 * H_ + c] + r * gh[gb + 2 * H_ + c]);
  float hv = (1.0f - z) * n + z * s[i];
  h[i]   = hv;
  hbf[i] = (__bf16)hv;
}

// ---------------------------------------------------------------------------
// r_out[b] = sigmoid(dot(h[b,:], Wr))  (O=1 -> plain f32 reduction)
// ---------------------------------------------------------------------------
__global__ __launch_bounds__(256) void k_rdot(const float* __restrict__ h,
                                              const float* __restrict__ Wr,
                                              float* __restrict__ out_r,
                                              int t)
{
  __shared__ float red[256];
  const int b = blockIdx.x;
  float p = 0.0f;
  for (int j = threadIdx.x; j < H_; j += 256)
    p += h[(size_t)b * H_ + j] * Wr[j];
  red[threadIdx.x] = p;
  __syncthreads();
  for (int sft = 128; sft > 0; sft >>= 1) {
    if (threadIdx.x < sft) red[threadIdx.x] += red[threadIdx.x + sft];
    __syncthreads();
  }
  if (threadIdx.x == 0) out_r[(size_t)b * T_ + t] = sigmoidf_(red[0]);
}

// ---------------------------------------------------------------------------
// fp32 -> bf16 conversion (8 elems/thread -> b128 traffic), and state init.
// ---------------------------------------------------------------------------
__global__ __launch_bounds__(256) void k_convert8(const float* __restrict__ src,
                                                  __bf16* __restrict__ dst, size_t n8)
{
  size_t i = (size_t)blockIdx.x * 256 + threadIdx.x;
  if (i < n8) {
    v8f v = *(const v8f*)(src + i * 8);
    v8bf o;
#pragma unroll
    for (int j = 0; j < 8; ++j) o[j] = (__bf16)v[j];
    *(v8bf*)(dst + i * 8) = o;
  }
}

__global__ __launch_bounds__(256) void k_init_s(const float* __restrict__ s_in,
                                                float* __restrict__ s_f,
                                                __bf16* __restrict__ s_bf)
{
  const int i = blockIdx.x * 256 + threadIdx.x;   // B*H threads
  float v = s_in[i];
  s_f[i]  = v;
  s_bf[i] = (__bf16)v;
}

// ---------------------------------------------------------------------------
extern "C" void kernel_launch(void* const* d_in, const int* in_sizes, int n_in,
                              void* d_out, int out_size, void* d_ws, size_t ws_size,
                              hipStream_t stream) {
  (void)in_sizes; (void)n_in; (void)out_size; (void)ws_size;

  const float* s_in = (const float*)d_in[0];   // [B,H]
  const float* a_in = (const float*)d_in[1];   // [B,T,I]
  const float* Wih0 = (const float*)d_in[2];   // [3H,I]
  /* d_in[3] = Whh0 : dead (layer-0 hidden is always 0) */
  const float* Wih1 = (const float*)d_in[4];   // [3H,H]
  const float* Whh1 = (const float*)d_in[5];   // [3H,H]
  const float* Wr   = (const float*)d_in[6];   // [1,H]
  const float* Wsl  = (const float*)d_in[7];   // [H,H]

  // Workspace layout (256B aligned slices)
  char* w = (char*)d_ws;
  size_t off = 0;
  auto alloc = [&](size_t bytes) -> void* {
    off = (off + 255) & ~(size_t)255;
    void* p = w + off;
    off += bytes;
    return p;
  };
  __bf16* a_bf    = (__bf16*)alloc((size_t)BT_ * I_ * 2);   // 16 MB
  __bf16* wih0_bf = (__bf16*)alloc((size_t)3 * H_ * I_ * 2);
  __bf16* wih1_bf = (__bf16*)alloc((size_t)G3_ * H_ * 2);
  __bf16* whh1_bf = (__bf16*)alloc((size_t)G3_ * H_ * 2);
  __bf16* ws_bf   = (__bf16*)alloc((size_t)H_ * H_ * 2);
  __bf16* h0_bf   = (__bf16*)alloc((size_t)BT_ * H_ * 2);   // 32 MB
  float*  gx1     = (float*) alloc((size_t)BT_ * G3_ * 4);  // 192 MB
  float*  gh      = (float*) alloc((size_t)B_ * G3_ * 4);
  float*  s_f     = (float*) alloc((size_t)B_ * H_ * 4);
  __bf16* s_bf    = (__bf16*)alloc((size_t)B_ * H_ * 2);
  float*  h_f     = (float*) alloc((size_t)B_ * H_ * 4);
  __bf16* h_bf    = (__bf16*)alloc((size_t)B_ * H_ * 2);

  float* out_r = (float*)d_out;            // [B,T,1]
  float* out_s = out_r + (size_t)B_ * T_;  // [B,T,H]

  auto cvt = [&](const float* src, __bf16* dst, size_t n) {
    size_t n8 = n / 8;   // all our sizes are multiples of 8
    k_convert8<<<dim3((unsigned)((n8 + 255) / 256)), dim3(256), 0, stream>>>(src, dst, n8);
  };
  cvt(a_in, a_bf, (size_t)BT_ * I_);
  cvt(Wih0, wih0_bf, (size_t)3 * H_ * I_);
  cvt(Wih1, wih1_bf, (size_t)G3_ * H_);
  cvt(Whh1, whh1_bf, (size_t)G3_ * H_);
  cvt(Wsl,  ws_bf,   (size_t)H_ * H_);
  k_init_s<<<dim3(B_ * H_ / 256), dim3(256), 0, stream>>>(s_in, s_f, s_bf);

  // Layer 0 (feedforward, fused gates) -> h0 bf16 [BT,H]   (fat 128x64 tiles)
  k_layer0<<<dim3(BT_ / 128, H_ / 64), dim3(256), 0, stream>>>(a_bf, wih0_bf, h0_bf);

  // gx1 = h0 @ Wih1^T  -> f32 [BT,3H]   (fat 128x64 tiles, 128x48 grid)
  k_gemm_f32<2><<<dim3(BT_ / 128, G3_ / 64), dim3(256), 0, stream>>>(
      h0_bf, wih1_bf, gx1, H_, H_, H_, G3_);

  // Recurrent scan over T (stream-ordered; Whh1/Ws stay hot in the 192MB L2).
  // Slim 64x64 tiles here: M=256, so block count (192/64) is what hides
  // latency across WGPs during the serial phase.
  for (int t = 0; t < T_; ++t) {
    k_gemm_f32<1><<<dim3(B_ / 64, G3_ / 64), dim3(256), 0, stream>>>(
        s_bf, whh1_bf, gh, H_, H_, H_, G3_);
    k_gates<<<dim3(B_ * H_ / 256), dim3(256), 0, stream>>>(gx1, gh, s_f, h_f, h_bf, t);
    k_sgemm_sig<<<dim3(B_ / 64, H_ / 64), dim3(256), 0, stream>>>(
        h_bf, ws_bf, s_f, s_bf, out_s, t);
    k_rdot<<<dim3(B_), dim3(256), 0, stream>>>(h_f, Wr, out_r, t);
  }
}
